// my_model_75393855914390
// MI455X (gfx1250) — matrix operand order repaired and checked by reading, not compiled, not run on hardware
//
#include <hip/hip_runtime.h>
#include <math.h>

typedef __attribute__((ext_vector_type(16))) __bf16        bf16x16;
typedef __attribute__((ext_vector_type(8)))  float         f32x8;
typedef __attribute__((ext_vector_type(4)))  float         f32x4;
typedef __attribute__((ext_vector_type(4)))  unsigned int  u32x4;

union FragU {
  bf16x16        v;
  u32x4          q[2];
  unsigned short s[16];
  __bf16         b[16];
};

// native bf16 convert (v_cvt_*bf16_f32), round-to-nearest-even
__device__ __forceinline__ unsigned short f2bf(float f) {
  __bf16 h = (__bf16)f;
  return __builtin_bit_cast(unsigned short, h);
}

// A fragment (16x32 bf16, MxK): lane<16 holds K{kb..kb+7, kb+16..kb+23}, kb = k + 8*(lane>=16)
__device__ __forceinline__ bf16x16 load_a_frag_f32(const float* __restrict__ rowp, int kb) {
  FragU a;
  f32x4 a0 = *(const f32x4*)(rowp + kb);
  f32x4 a1 = *(const f32x4*)(rowp + kb + 4);
  f32x4 a2 = *(const f32x4*)(rowp + kb + 16);
  f32x4 a3 = *(const f32x4*)(rowp + kb + 20);
  a.b[0]  = (__bf16)a0.x; a.b[1]  = (__bf16)a0.y; a.b[2]  = (__bf16)a0.z; a.b[3]  = (__bf16)a0.w;
  a.b[4]  = (__bf16)a1.x; a.b[5]  = (__bf16)a1.y; a.b[6]  = (__bf16)a1.z; a.b[7]  = (__bf16)a1.w;
  a.b[8]  = (__bf16)a2.x; a.b[9]  = (__bf16)a2.y; a.b[10] = (__bf16)a2.z; a.b[11] = (__bf16)a2.w;
  a.b[12] = (__bf16)a3.x; a.b[13] = (__bf16)a3.y; a.b[14] = (__bf16)a3.z; a.b[15] = (__bf16)a3.w;
  return a.v;
}

__device__ __forceinline__ bf16x16 load_a_frag_bf16(const unsigned short* __restrict__ rowp, int kb) {
  FragU a;
  a.q[0] = *(const u32x4*)(rowp + kb);
  a.q[1] = *(const u32x4*)(rowp + kb + 16);
  return a.v;
}

// ---------------- big GEMM:  C[M,N] = A[M,K](f32) @ Bt[N,K](bf16)^T + bias ----------------
__global__ __launch_bounds__(256) void gemm_bias_kernel(
    const float* __restrict__ A, const unsigned short* __restrict__ Bt,
    const float* __restrict__ bias, float* __restrict__ C, int M, int K, int N) {
  const int lane = threadIdx.x & 31;
  const int wave = threadIdx.x >> 5;
  const int lo = lane & 15, hi = lane >> 4;
  const int r0 = blockIdx.x * 64 + (wave >> 1) * 16;
  const int c0 = blockIdx.y * 128 + (wave & 1) * 64;
  const int arow = (r0 + lo < M) ? (r0 + lo) : (M - 1);
  const float* rowp = A + (size_t)arow * K;
  f32x8 acc[4];
#pragma unroll
  for (int t = 0; t < 4; ++t)
#pragma unroll
    for (int v = 0; v < 8; ++v) acc[t][v] = 0.f;
#pragma unroll 2
  for (int k = 0; k < K; k += 32) {
    const int kbB = k + hi * 8;
    // issue all B loads first so WMMAs can drain with staggered waitcnts
    FragU braw[4];
#pragma unroll
    for (int t = 0; t < 4; ++t) {
      const unsigned short* colp = Bt + (size_t)(c0 + t * 16 + lo) * K + kbB;
      braw[t].q[0] = *(const u32x4*)(colp);
      braw[t].q[1] = *(const u32x4*)(colp + 16);
    }
    // A load + cvt overlaps B load latency
    bf16x16 af = load_a_frag_f32(rowp, k + hi * 8);
#pragma unroll
    for (int t = 0; t < 4; ++t)
      acc[t] = __builtin_amdgcn_wmma_f32_16x16x32_bf16(false, af, false, braw[t].v,
                                                       (short)0, acc[t], false, false);
  }
#pragma unroll
  for (int t = 0; t < 4; ++t) {
    const int col = c0 + t * 16 + lo;
    const float bv = bias[col];
#pragma unroll
    for (int v = 0; v < 8; ++v) {
      const int row = r0 + v + hi * 8;
      if (row < M) C[(size_t)row * N + col] = acc[t][v] + bv;
    }
  }
}

// ---------------- row L2 normalize (H=512), optional bf16 copy ----------------
__global__ __launch_bounds__(128) void l2norm_kernel(float* __restrict__ X,
                                                     unsigned short* __restrict__ Xh, int H) {
  __shared__ float red[128];
  const int row = blockIdx.x;
  const int t = threadIdx.x;
  float* xr = X + (size_t)row * H;
  float v[4];
  float s = 0.f;
#pragma unroll
  for (int i = 0; i < 4; ++i) { v[i] = xr[t + i * 128]; s += v[i] * v[i]; }
  red[t] = s;
  __syncthreads();
  for (int off = 64; off > 0; off >>= 1) {
    if (t < off) red[t] += red[t + off];
    __syncthreads();
  }
  const float scale = 1.f / fmaxf(sqrtf(red[0]), 1e-12f);
#pragma unroll
  for (int i = 0; i < 4; ++i) {
    const float o = v[i] * scale;
    xr[t + i * 128] = o;
    if (Xh) Xh[(size_t)row * H + t + i * 128] = f2bf(o);
  }
}

// ---------------- fused logits (K=512, N=64) + softmax ----------------
__global__ __launch_bounds__(256) void logits_softmax_kernel(
    const float* __restrict__ O, const unsigned short* __restrict__ Wct,
    const float* __restrict__ bias, float* __restrict__ U, int M) {
  const int K = 512;
  __shared__ float lg[64][64];
  const int lane = threadIdx.x & 31;
  const int wave = threadIdx.x >> 5;
  const int lo = lane & 15, hi = lane >> 4;
  const int rl0 = (wave >> 1) * 16;
  const int cl0 = (wave & 1) * 32;
  const int r0 = blockIdx.x * 64 + rl0;
  const int arow = (r0 + lo < M) ? (r0 + lo) : (M - 1);
  const float* rowp = O + (size_t)arow * K;
  f32x8 acc[2];
#pragma unroll
  for (int t = 0; t < 2; ++t)
#pragma unroll
    for (int v = 0; v < 8; ++v) acc[t][v] = 0.f;
#pragma unroll 2
  for (int k = 0; k < K; k += 32) {
    const int kbB = k + hi * 8;
    FragU braw[2];
#pragma unroll
    for (int t = 0; t < 2; ++t) {
      const unsigned short* colp = Wct + (size_t)(cl0 + t * 16 + lo) * K + kbB;
      braw[t].q[0] = *(const u32x4*)(colp);
      braw[t].q[1] = *(const u32x4*)(colp + 16);
    }
    bf16x16 af = load_a_frag_f32(rowp, k + hi * 8);
#pragma unroll
    for (int t = 0; t < 2; ++t)
      acc[t] = __builtin_amdgcn_wmma_f32_16x16x32_bf16(false, af, false, braw[t].v,
                                                       (short)0, acc[t], false, false);
  }
#pragma unroll
  for (int t = 0; t < 2; ++t) {
    const int col = cl0 + t * 16 + lo;
    const float bv = bias[col];
#pragma unroll
    for (int v = 0; v < 8; ++v) lg[rl0 + v + hi * 8][col] = acc[t][v] + bv;
  }
  __syncthreads();
  const int t = threadIdx.x;
  if (t < 64) {
    const int row = blockIdx.x * 64 + t;
    if (row < M) {
      float m = lg[t][0];
      for (int j = 1; j < 64; ++j) m = fmaxf(m, lg[t][j]);
      float s = 0.f;
      for (int j = 0; j < 64; ++j) { const float e = __expf(lg[t][j] - m); lg[t][j] = e; s += e; }
      const float inv = 1.f / s;
      float* up = U + (size_t)row * 64;
      for (int j = 0; j < 64; ++j) up[j] = lg[t][j] * inv;
    }
  }
}

// ---------------- kmeans: distances (K=512, 64 centers) + argmin ----------------
__global__ __launch_bounds__(256) void kmeans_label_kernel(
    const unsigned short* __restrict__ Xh, const unsigned short* __restrict__ Ch,
    const float* __restrict__ csq, int* __restrict__ labels, int M) {
  const int K = 512;
  __shared__ float dd[64][64];
  const int lane = threadIdx.x & 31;
  const int wave = threadIdx.x >> 5;
  const int lo = lane & 15, hi = lane >> 4;
  const int rl0 = (wave >> 1) * 16;
  const int cl0 = (wave & 1) * 32;
  const int r0 = blockIdx.x * 64 + rl0;
  const int arow = (r0 + lo < M) ? (r0 + lo) : (M - 1);
  const unsigned short* rowp = Xh + (size_t)arow * K;
  f32x8 acc[2];
#pragma unroll
  for (int t = 0; t < 2; ++t)
#pragma unroll
    for (int v = 0; v < 8; ++v) acc[t][v] = 0.f;
#pragma unroll 2
  for (int k = 0; k < K; k += 32) {
    const int kbB = k + hi * 8;
    FragU braw[2];
#pragma unroll
    for (int t = 0; t < 2; ++t) {
      const unsigned short* colp = Ch + (size_t)(cl0 + t * 16 + lo) * K + kbB;
      braw[t].q[0] = *(const u32x4*)(colp);
      braw[t].q[1] = *(const u32x4*)(colp + 16);
    }
    bf16x16 af = load_a_frag_bf16(rowp, k + hi * 8);
#pragma unroll
    for (int t = 0; t < 2; ++t)
      acc[t] = __builtin_amdgcn_wmma_f32_16x16x32_bf16(false, af, false, braw[t].v,
                                                       (short)0, acc[t], false, false);
  }
#pragma unroll
  for (int t = 0; t < 2; ++t) {
    const int col = cl0 + t * 16 + lo;
    const float cq = csq[col];
#pragma unroll
    for (int v = 0; v < 8; ++v) dd[rl0 + v + hi * 8][col] = cq - 2.f * acc[t][v];
  }
  __syncthreads();
  const int t = threadIdx.x;
  if (t < 64) {
    const int row = blockIdx.x * 64 + t;
    if (row < M) {
      int best = 0;
      float bv = dd[t][0];
      for (int j = 1; j < 64; ++j)
        if (dd[t][j] < bv) { bv = dd[t][j]; best = j; }
      labels[row] = best;
    }
  }
}

// ---------------- kmeans: LDS-staged segment sums (64 clusters x 512 dims) ----------------
__global__ __launch_bounds__(256) void kmeans_accum_kernel(
    const float* __restrict__ X, const int* __restrict__ labels,
    float* __restrict__ sums, float* __restrict__ counts, int M) {
  extern __shared__ float ls[];   // 64*512 floats = 128 KB
  __shared__ int slab[128];
  const int t = threadIdx.x;
  for (int i = t; i < 64 * 512; i += 256) ls[i] = 0.f;
  const int p0 = blockIdx.x * 128;
  const int npts = (M - p0 < 128) ? (M - p0) : 128;
  if (t < npts) slab[t] = labels[p0 + t];
  __syncthreads();
  for (int p = 0; p < npts; ++p) {
    float* sr = ls + (size_t)slab[p] * 512;
    const float* xr = X + (size_t)(p0 + p) * 512;
    sr[t] += xr[t];
    sr[t + 256] += xr[t + 256];
  }
  __syncthreads();
  for (int i = t; i < 64 * 512; i += 256) {
    const float v = ls[i];
    if (v != 0.f) atomicAdd(&sums[i], v);
  }
  if (t < 64) {
    float c = 0.f;
    for (int p = 0; p < npts; ++p) c += (slab[p] == t) ? 1.f : 0.f;
    if (c > 0.f) atomicAdd(&counts[t], c);
  }
}

__global__ __launch_bounds__(256) void kmeans_update_kernel(
    const float* __restrict__ sums, const float* __restrict__ counts,
    float* __restrict__ cent, unsigned short* __restrict__ centh) {
  const int idx = blockIdx.x * 256 + threadIdx.x;
  if (idx >= 64 * 512) return;
  const float c = counts[idx >> 9];
  const float v = (c > 0.f) ? (sums[idx] / c) : cent[idx];
  cent[idx] = v;
  centh[idx] = f2bf(v);
}

__global__ void csq_kernel(const float* __restrict__ cent, float* __restrict__ csq) {
  const int j = threadIdx.x;
  if (j >= 64) return;
  const float* cr = cent + (size_t)j * 512;
  float s = 0.f;
  for (int k = 0; k < 512; ++k) s += cr[k] * cr[k];
  csq[j] = s;
}

__global__ __launch_bounds__(256) void init_centers_kernel(
    const float* __restrict__ src, float* __restrict__ cent, unsigned short* __restrict__ centh) {
  const int idx = blockIdx.x * 256 + threadIdx.x;
  if (idx >= 64 * 512) return;
  const float v = src[idx];
  cent[idx] = v;
  centh[idx] = f2bf(v);
}

// in[Kd][Nd] f32 -> out[Nd][Kd] bf16
__global__ __launch_bounds__(256) void transpose_bf16_kernel(
    const float* __restrict__ in, unsigned short* __restrict__ out, int Kd, int Nd) {
  const int idx = blockIdx.x * 256 + threadIdx.x;
  if (idx >= Kd * Nd) return;
  const int n = idx / Kd, k = idx - n * Kd;
  out[idx] = f2bf(in[(size_t)k * Nd + n]);
}

extern "C" void kernel_launch(void* const* d_in, const int* in_sizes, int n_in,
                              void* d_out, int out_size, void* d_ws, size_t ws_size,
                              hipStream_t stream) {
  (void)n_in; (void)out_size; (void)ws_size;
  const float* x  = (const float*)d_in[0];
  const float* xw = (const float*)d_in[1];
  const float* W1 = (const float*)d_in[2];
  const float* b1 = (const float*)d_in[3];
  const float* W2 = (const float*)d_in[4];
  const float* b2 = (const float*)d_in[5];
  const float* Wc = (const float*)d_in[6];
  const float* bc = (const float*)d_in[7];

  const int H  = in_sizes[3];        // 512
  const int D  = in_sizes[2] / H;    // 1024
  const int M  = in_sizes[0] / D;    // 50000
  const int Kc = in_sizes[7];        // 64

  // d_out layout: out1 | out2 | labels(int32) | centers | u1 | u2
  float* outf = (float*)d_out;
  float* out1 = outf;
  float* out2 = out1 + (size_t)M * H;
  int*   labl = (int*)(out2 + (size_t)M * H);
  float* cout = (float*)(labl + M);
  float* u1   = cout + (size_t)Kc * H;
  float* u2   = u1 + (size_t)M * Kc;

  char* w = (char*)d_ws;
  auto alloc = [&](size_t bytes) { char* p = w; w += (bytes + 255) & ~(size_t)255; return p; };
  unsigned short* W1t   = (unsigned short*)alloc((size_t)D * H * 2);
  unsigned short* W2t   = (unsigned short*)alloc((size_t)D * H * 2);
  unsigned short* Wct   = (unsigned short*)alloc((size_t)H * Kc * 2);
  unsigned short* out2h = (unsigned short*)alloc((size_t)M * H * 2);
  unsigned short* centh = (unsigned short*)alloc((size_t)Kc * H * 2);
  float* cent   = (float*)alloc((size_t)Kc * H * 4);
  float* sums   = (float*)alloc((size_t)Kc * H * 4 + (size_t)Kc * 4);  // counts appended
  float* counts = sums + (size_t)Kc * H;
  float* csq    = (float*)alloc((size_t)Kc * 4);

  hipFuncSetAttribute((const void*)kmeans_accum_kernel,
                      hipFuncAttributeMaxDynamicSharedMemorySize, 64 * 512 * 4);

  dim3 blk(256);
  transpose_bf16_kernel<<<(D * H + 255) / 256, blk, 0, stream>>>(W1, W1t, D, H);
  transpose_bf16_kernel<<<(D * H + 255) / 256, blk, 0, stream>>>(W2, W2t, D, H);
  transpose_bf16_kernel<<<(H * Kc + 255) / 256, blk, 0, stream>>>(Wc, Wct, H, Kc);

  const int MB = (M + 63) / 64;
  gemm_bias_kernel<<<dim3(MB, H / 128), blk, 0, stream>>>(x,  W1t, b1, out1, M, D, H);
  gemm_bias_kernel<<<dim3(MB, H / 128), blk, 0, stream>>>(xw, W2t, b2, out2, M, D, H);
  l2norm_kernel<<<M, 128, 0, stream>>>(out1, (unsigned short*)nullptr, H);
  l2norm_kernel<<<M, 128, 0, stream>>>(out2, out2h, H);
  logits_softmax_kernel<<<MB, blk, 0, stream>>>(out1, Wct, bc, u1, M);
  logits_softmax_kernel<<<MB, blk, 0, stream>>>(out2, Wct, bc, u2, M);

  init_centers_kernel<<<(Kc * H + 255) / 256, blk, 0, stream>>>(out2, cent, centh);
  csq_kernel<<<1, 64, 0, stream>>>(cent, csq);
  const int AB = (M + 127) / 128;
  for (int it = 0; it < 25; ++it) {
    hipMemsetAsync(sums, 0, (size_t)Kc * H * 4 + (size_t)Kc * 4, stream);
    kmeans_label_kernel<<<MB, blk, 0, stream>>>(out2h, centh, csq, labl, M);
    kmeans_accum_kernel<<<AB, blk, 64 * 512 * 4, stream>>>(out2, labl, sums, counts, M);
    kmeans_update_kernel<<<(Kc * H + 255) / 256, blk, 0, stream>>>(sums, counts, cent, centh);
    csq_kernel<<<1, 64, 0, stream>>>(cent, csq);
  }
  kmeans_label_kernel<<<MB, blk, 0, stream>>>(out2h, centh, csq, labl, M);
  hipMemcpyAsync(cout, cent, (size_t)Kc * H * 4, hipMemcpyDeviceToDevice, stream);
}
